// MultiHeadedAttention_39951785787979
// MI455X (gfx1250) — matrix operand order repaired and checked by reading, not compiled, not run on hardware
//
#include <hip/hip_runtime.h>
#include <hip/hip_bf16.h>
#include <stdint.h>

// MHA fwd: B=4, S=1024, U=1024, H=16, dk=64. All GEMMs via v_wmma_f32_16x16x32_bf16.
#define BB   4
#define SS   1024
#define UU   1024
#define HH   16
#define DK   64
#define MTOT (BB*SS)        // 4096 tokens

typedef __attribute__((ext_vector_type(16))) __bf16 v16bf;
typedef __attribute__((ext_vector_type(8)))  float  v8f;

union FragAB { uint32_t u[8]; uint4 q[2]; __bf16 h[16]; v16bf v; };

// native f32->bf16 (backend RNE cvt; ideally fuses to v_cvt_pk_bf16_f32)
__device__ __forceinline__ uint32_t packbf(float x, float y) {
  union { __bf16 h[2]; uint32_t u; } r;
  r.h[0] = (__bf16)x; r.h[1] = (__bf16)y;
  return r.u;
}
__device__ __forceinline__ __bf16 f2bf(float f) { return (__bf16)f; }

__device__ __forceinline__ v8f wmma_bf16(const FragAB& a, const FragAB& b, v8f c) {
  return __builtin_amdgcn_wmma_f32_16x16x32_bf16(false, a.v, false, b.v, (short)0, c, false, false);
}

// ---- Wt[k*U + n] = bf16(W[n*U + k]); each thread packs 2 cols -> coalesced b32 stores
__global__ void wconv_kernel(const float* __restrict__ W, __bf16* __restrict__ Wt) {
  int id = blockIdx.x * blockDim.x + threadIdx.x;          // 0 .. U*U/2-1
  int k = id >> 9, n = (id & 511) * 2;
  uint32_t p = packbf(W[(size_t)n * UU + k], W[(size_t)(n + 1) * UU + k]);
  *(uint32_t*)(Wt + (size_t)k * UU + n) = p;
}

// ---- GEMM: out[m, n] = sum_k X[m,k]*Wt[k,n] + bias[n]; wave tile = 16 rows x 64 cols
// (64-col tile == one head for modes 0/1). A-frag reused across 4 WMMAs per K-step.
// mode 0: bf16 transposed per head Qt[b,h,d,s], scaled 1/64 (scores/dk folded in)
// mode 1: bf16 row-major per head KV[b,h,s,d]
// mode 2: f32 flat out[m,n]
__global__ void gemm_kernel(const float* __restrict__ X, const __bf16* __restrict__ Wt,
                            const float* __restrict__ bias, void* __restrict__ outp, int mode) {
  int gw   = (blockIdx.x * blockDim.x + threadIdx.x) >> 5;   // 0..4095
  int lane = threadIdx.x & 31;
  int l = lane & 15, hi = lane >> 4;
  int tn4 = gw & 15;                 // 64-col tile == head index
  int tm  = gw >> 4;                 // 0..255
  int m = tm * 16 + l;               // A-frag row (lane <-> M)
  const float* xrow = X + (size_t)m * UU;
  v8f acc[4] = {{}, {}, {}, {}};
  for (int kk = 0; kk < UU / 32; ++kk) {
    FragAB a;
    int base = hi * 8 + kk * 32;     // A: lane=M; VGPR->K per ISA 16-bit A layout
    float4 f0 = *(const float4*)(xrow + base);
    float4 f1 = *(const float4*)(xrow + base + 4);
    float4 f2 = *(const float4*)(xrow + base + 16);
    float4 f3 = *(const float4*)(xrow + base + 20);
    a.u[0] = packbf(f0.x, f0.y); a.u[1] = packbf(f0.z, f0.w);
    a.u[2] = packbf(f1.x, f1.y); a.u[3] = packbf(f1.z, f1.w);
    a.u[4] = packbf(f2.x, f2.y); a.u[5] = packbf(f2.z, f2.w);
    a.u[6] = packbf(f3.x, f3.y); a.u[7] = packbf(f3.z, f3.w);
    int kd = l + hi * 16 + kk * 32;  // B: lane=N, VGPR->K (kd, wrow unused)
    const __bf16* wrow = Wt + (size_t)kd * UU + tn4 * 64;
#pragma unroll
    for (int nt = 0; nt < 4; ++nt) {
      FragAB bfr;
#pragma unroll
      for (int s = 0; s < 16; ++s)
        bfr.h[s] = Wt[(size_t)(kk * 32 + hi * 8 + (s & 7) + (s >> 3) * 16) * UU + tn4 * 64 + nt * 16 + l];
      acc[nt] = wmma_bf16(a, bfr, acc[nt]);
    }
  }
  int b  = (tm * 16) >> 10;          // 16-row tile never straddles a batch
  int bh = b * HH + tn4;
#pragma unroll
  for (int nt = 0; nt < 4; ++nt) {
    int ncol = tn4 * 64 + nt * 16 + l;
    float bv = bias[ncol];
#pragma unroll
    for (int v = 0; v < 8; ++v) acc[nt][v] += bv;
    if (mode == 2) {
      float* O = (float*)outp;
#pragma unroll
      for (int v = 0; v < 8; ++v)
        O[(size_t)(tm * 16 + v + hi * 8) * UU + ncol] = acc[nt][v];
    } else {
      __bf16* O = (__bf16*)outp;
      int d = nt * 16 + l;
#pragma unroll
      for (int v = 0; v < 8; ++v) {
        int s = (tm * 16 + v + hi * 8) & (SS - 1);
        if (mode == 1) O[((size_t)bh * SS + s) * DK + d] = f2bf(acc[nt][v]);
        else           O[((size_t)bh * DK + d) * SS + s] = f2bf(acc[nt][v] * (1.0f / 64.0f));
      }
    }
  }
}

// ---- Flash attention: one wave = 16 query rows x dk=64, streaming 32 keys/step.
// scoresT = K_tile x Q^T so the D-frag (lane<->q, VGPR<->k) repacks to the P
// A-frag (lane<->q, VGPR<->k) with per-lane register moves only.
__global__ void attn_kernel(const __bf16* __restrict__ Qt, const __bf16* __restrict__ Kb,
                            const __bf16* __restrict__ Vb, const int* __restrict__ mask,
                            float* __restrict__ attnF) {
  int gw   = (blockIdx.x * blockDim.x + threadIdx.x) >> 5;  // 0..4095
  int lane = threadIdx.x & 31;
  int l = lane & 15, hi = lane >> 4;
  int qt = gw & 63;
  int bh = gw >> 6;
  int b = bh >> 4, hidx = bh & 15;
  int q = qt * 16 + l;

  FragAB qb[2];                         // Q^T B-frags, invariant over key loop
#pragma unroll
  for (int ds = 0; ds < 2; ++ds) {
    int kd = l + hi * 16 + ds * 32;     // d index (lane <-> K-dim)
    const __bf16* qrow = Qt + ((size_t)bh * DK + kd) * SS + qt * 16;
#pragma unroll
    for (int s = 0; s < 16; ++s)
      qb[ds].h[s] = Qt[((size_t)bh * DK + ds * 32 + hi * 8 + (s & 7) + (s >> 3) * 16) * SS + q];
  }

  v8f o[4] = {{}, {}, {}, {}};
  float mrow = -__builtin_inff(), lrow = 0.f;
  const int* mptr = mask + ((size_t)b * SS + q) * SS;

  for (int ks = 0; ks < SS / 32; ++ks) {
    const __bf16* krp = Kb + ((size_t)bh * SS + ks * 32 + l) * DK;
    const __bf16* vrp = Vb + ((size_t)bh * SS + ks * 32 + l + hi * 16) * DK;
    if (ks + 1 < SS / 32) {             // hide next tile behind the WMMA chain
      __builtin_prefetch(krp + 32 * DK, 0, 0);
      __builtin_prefetch(vrp + 32 * DK, 0, 0);
    }
    v8f t0 = {}, t1 = {};
#pragma unroll
    for (int ds = 0; ds < 2; ++ds) {
      FragAB a;
      int kb = hi * 8 + ds * 32;        // A VGPR->K runs are 8 consecutive bf16
      a.q[0] = *(const uint4*)(krp + kb);
      a.q[1] = *(const uint4*)(krp + kb + 16);
      t0 = wmma_bf16(a, qb[ds], t0);
      const __bf16* krp1 = krp + (size_t)16 * DK;
      a.q[0] = *(const uint4*)(krp1 + kb);
      a.q[1] = *(const uint4*)(krp1 + kb + 16);
      t1 = wmma_bf16(a, qb[ds], t1);
    }
    // causal mask (read faithfully): t[k,q], k = v + hi*8 (+16 for t1)
    const int* mk = mptr + ks * 32 + hi * 8;
    int4 m0 = *(const int4*)(mk);
    int4 m1 = *(const int4*)(mk + 4);
    int4 m2 = *(const int4*)(mk + 16);
    int4 m3 = *(const int4*)(mk + 20);
    if (m0.x == 0) t0[0] = -1e9f;  if (m0.y == 0) t0[1] = -1e9f;
    if (m0.z == 0) t0[2] = -1e9f;  if (m0.w == 0) t0[3] = -1e9f;
    if (m1.x == 0) t0[4] = -1e9f;  if (m1.y == 0) t0[5] = -1e9f;
    if (m1.z == 0) t0[6] = -1e9f;  if (m1.w == 0) t0[7] = -1e9f;
    if (m2.x == 0) t1[0] = -1e9f;  if (m2.y == 0) t1[1] = -1e9f;
    if (m2.z == 0) t1[2] = -1e9f;  if (m2.w == 0) t1[3] = -1e9f;
    if (m3.x == 0) t1[4] = -1e9f;  if (m3.y == 0) t1[5] = -1e9f;
    if (m3.z == 0) t1[6] = -1e9f;  if (m3.w == 0) t1[7] = -1e9f;
    // online softmax: row q = lane&15; other 16 values live in lane^16
    float loc = -__builtin_inff();
#pragma unroll
    for (int v = 0; v < 8; ++v) loc = fmaxf(loc, fmaxf(t0[v], t1[v]));
    loc = fmaxf(loc, __shfl_xor(loc, 16, 32));
    float mnew  = fmaxf(mrow, loc);
    float alpha = __expf(mrow - mnew);
    float p0[8], p1[8], psum = 0.f;
#pragma unroll
    for (int v = 0; v < 8; ++v) {
      p0[v] = __expf(t0[v] - mnew);
      p1[v] = __expf(t1[v] - mnew);
      psum += p0[v] + p1[v];
    }
    psum += __shfl_xor(psum, 16, 32);
    lrow = lrow * alpha + psum;
    mrow = mnew;
    // repack P (16x32) into A-frag: pure per-lane packs (layouts line up)
    FragAB pa;
#pragma unroll
    for (int v2 = 0; v2 < 8; ++v2) {
      int i0 = 2 * (v2 & 3);
      pa.u[v2] = (v2 < 4) ? packbf(p0[i0], p0[i0 + 1]) : packbf(p1[i0], p1[i0 + 1]);
    }
    // rescale O accumulators: O-frag row q = v + 8*hi -> fetch alpha from lane q
    float ar[8];
#pragma unroll
    for (int v = 0; v < 8; ++v) ar[v] = __shfl(alpha, v + hi * 8, 32);
#pragma unroll
    for (int v = 0; v < 8; ++v) {
      o[0][v] *= ar[v]; o[1][v] *= ar[v]; o[2][v] *= ar[v]; o[3][v] *= ar[v];
    }
    // O += P x V  (V B-frags: lane=d, VGPR=k)
#pragma unroll
    for (int t4 = 0; t4 < 4; ++t4) {
      FragAB vb0;
#pragma unroll
      for (int s = 0; s < 16; ++s)
        vb0.h[s] = Vb[((size_t)bh * SS + ks * 32 + hi * 8 + (s & 7) + (s >> 3) * 16) * DK + t4 * 16 + l];
      o[t4] = wmma_bf16(pa, vb0, o[t4]);
    }
  }
  // normalize and write attn output (f32, [b,s, h*64+d]) for the final projection
  float linv = 1.0f / lrow;
  float lr[8];
#pragma unroll
  for (int v = 0; v < 8; ++v) lr[v] = __shfl(linv, v + hi * 8, 32);
#pragma unroll
  for (int t4 = 0; t4 < 4; ++t4) {
#pragma unroll
    for (int v = 0; v < 8; ++v) {
      size_t row = (size_t)b * SS + qt * 16 + v + hi * 8;
      attnF[row * UU + hidx * DK + t4 * 16 + l] = o[t4][v] * lr[v];
    }
  }
}

extern "C" void kernel_launch(void* const* d_in, const int* in_sizes, int n_in,
                              void* d_out, int out_size, void* d_ws, size_t ws_size,
                              hipStream_t stream) {
  (void)in_sizes; (void)n_in; (void)out_size; (void)ws_size;
  const float* query = (const float*)d_in[0];
  const float* key   = (const float*)d_in[1];
  const float* value = (const float*)d_in[2];
  const int*   mask  = (const int*)  d_in[3];
  const float* Wq = (const float*)d_in[4];
  const float* bq = (const float*)d_in[5];
  const float* Wk = (const float*)d_in[6];
  const float* bk = (const float*)d_in[7];
  const float* Wv = (const float*)d_in[8];
  const float* bv = (const float*)d_in[9];
  const float* Wo = (const float*)d_in[10];
  const float* bo = (const float*)d_in[11];

  // workspace layout (48 MB total)
  char* ws = (char*)d_ws;
  const size_t WSZ = (size_t)UU * UU;        // 1M elems per weight
  const size_t QSZ = (size_t)MTOT * UU;      // 4M elems per activation
  __bf16* Wt_q = (__bf16*)ws;
  __bf16* Wt_k = Wt_q + WSZ;
  __bf16* Wt_v = Wt_k + WSZ;
  __bf16* Wt_o = Wt_v + WSZ;
  __bf16* Qt   = Wt_o + WSZ;                 // [B,H,DK,S] bf16, pre-scaled 1/64
  __bf16* Kb   = Qt + QSZ;                   // [B,H,S,DK] bf16
  __bf16* Vb   = Kb + QSZ;                   // [B,H,S,DK] bf16
  float*  attnF = (float*)(Vb + QSZ);        // [B*S, U] f32

  // 1) weight transpose+convert (2 packed cols/thread)
  dim3 wblk(256), wgrd((UU * UU / 2) / 256);
  hipLaunchKernelGGL(wconv_kernel, wgrd, wblk, 0, stream, Wq, Wt_q);
  hipLaunchKernelGGL(wconv_kernel, wgrd, wblk, 0, stream, Wk, Wt_k);
  hipLaunchKernelGGL(wconv_kernel, wgrd, wblk, 0, stream, Wv, Wt_v);
  hipLaunchKernelGGL(wconv_kernel, wgrd, wblk, 0, stream, Wo, Wt_o);

  // 2) projections: 4096 wave-tiles (16x64), 4 waves (128 thr) per block
  dim3 gblk(128), ggrd(1024);
  hipLaunchKernelGGL(gemm_kernel, ggrd, gblk, 0, stream, query, Wt_q, bq, (void*)Qt, 0);
  hipLaunchKernelGGL(gemm_kernel, ggrd, gblk, 0, stream, key,   Wt_k, bk, (void*)Kb, 1);
  hipLaunchKernelGGL(gemm_kernel, ggrd, gblk, 0, stream, value, Wt_v, bv, (void*)Vb, 1);

  // 3) flash attention: 4096 waves
  hipLaunchKernelGGL(attn_kernel, dim3(1024), dim3(128), 0, stream, Qt, Kb, Vb, mask, attnF);

  // 4) output projection -> d_out (f32)
  hipLaunchKernelGGL(gemm_kernel, ggrd, gblk, 0, stream, attnF, Wt_o, bo, d_out, 2);
}
